// GCNSampling_91302414778942
// MI455X (gfx1250) — matrix operand order repaired and checked by reading, not compiled, not run on hardware
//
#include <hip/hip_runtime.h>

// ---------------------------------------------------------------------------
// GCN sampling pipeline for MI455X (gfx1250, wave32, WMMA bf16).
//   agg0 (mem-bound f32 gather, writes WMMA A-fragment-packed bf16)
//   gemm1 (v_wmma_f32_16x16x32_bf16, 2x4 tile register blocking, bias+ReLU)
//   agg1 (bf16 gather, A-fragment-packed bf16 out)
//   gemm2 (v_wmma_f32_16x16x32_bf16, 1x4 tile register blocking, bias)
// ---------------------------------------------------------------------------

typedef __bf16 v16bf __attribute__((ext_vector_type(16)));
typedef __bf16 v8bf  __attribute__((ext_vector_type(8)));
typedef float  v8f   __attribute__((ext_vector_type(8)));

#define GN0    409600
#define GN1    40960
#define GN2    4096
#define FANOUT 10
#define IN_F   512
#define HID    256
#define NCLS   64

#define KT1 (IN_F / 32)   // 16 K-tiles for GEMM1
#define NT1 (HID / 16)    // 16 N-tiles for GEMM1
#define KT2 (HID / 32)    // 8  K-tiles for GEMM2
#define NT2 (NCLS / 16)   // 4  N-tiles for GEMM2

// A-fragment packing (16-bit A 16x32, ISA 7.12.2):
//   lane = row + 16*(K in {8..15,24..31}),  e = (K&7) + 8*(K>=16)
// B-fragment packing (16-bit B 32x16):
//   lane = n + 16*(K in {8..15,24..31}),    e = (K&7) + 8*(K>=16)
// each tile occupies 512 consecutive bf16 (32 lanes x 16 elems, 1 KB).

// ---------------- aggregation block 0: f32 gather -> Apack bf16 -------------
__global__ __launch_bounds__(256) void k_agg0(const float* __restrict__ feat,
                                              const int* __restrict__ src0,
                                              __bf16* __restrict__ Apack) {
  const int t = threadIdx.x;
  const int rl = t >> 6;                 // 4 rows per block
  const int g  = t & 63;                 // 64 groups of 8 columns
  const int d  = blockIdx.x * 4 + rl;    // dst node
  const int* s = src0 + d * FANOUT;

  float acc[8];
#pragma unroll
  for (int i = 0; i < 8; ++i) acc[i] = 0.f;
#pragma unroll
  for (int j = 0; j < FANOUT; ++j) {
    const float* r = feat + (size_t)s[j] * IN_F + g * 8;
    const float4 a = *(const float4*)(r);
    const float4 b = *(const float4*)(r + 4);
    acc[0] += a.x; acc[1] += a.y; acc[2] += a.z; acc[3] += a.w;
    acc[4] += b.x; acc[5] += b.y; acc[6] += b.z; acc[7] += b.w;
  }
  v8bf o;
#pragma unroll
  for (int i = 0; i < 8; ++i) o[i] = (__bf16)(acc[i] * 0.1f);

  const int mT  = d >> 4, row = d & 15;
  const int kT  = g >> 2;
  const int kin = (g & 3) * 8;           // 0,8,16,24
  const int lane = row + ((kin & 8) ? 16 : 0);
  const int e    = (kin & 16) ? 8 : 0;
  *(v8bf*)(Apack + ((size_t)(mT * KT1 + kT) * 512 + lane * 16 + e)) = o;
}

// ------------- pack weight matrix [K][N] f32 -> B-fragment bf16 -------------
__global__ void k_packW(const float* __restrict__ W, __bf16* __restrict__ Wp,
                        int K, int N) {
  const int idx = blockIdx.x * blockDim.x + threadIdx.x;
  if (idx >= K * N) return;
  const int k = idx / N, n = idx % N;
  const int kT = k >> 5, kin = k & 31;
  const int nT = n >> 4, nin = n & 15;
  const int lane = nin + ((kin & 8) ? 16 : 0);
  const int e    = (kin & 7) + ((kin & 16) ? 8 : 0);
  Wp[((size_t)(kT * (N >> 4) + nT) * 512) + lane * 16 + e] = (__bf16)W[idx];
}

// --------- GEMM1: relu(Apack x W1 + b1) -> H bf16 (2x4 tile blocking) -------
__global__ __launch_bounds__(256) void k_gemm1(const __bf16* __restrict__ Apack,
                                               const __bf16* __restrict__ Bpack,
                                               const float* __restrict__ bias,
                                               __bf16* __restrict__ H) {
  const int lane = threadIdx.x & 31;
  const int wid  = threadIdx.x >> 5;
  const int grp  = blockIdx.x * 8 + wid;  // 5120 wave-groups
  const int nG = grp & 3;                 // 4 N-groups (4 tiles each)
  const int mG = grp >> 2;                // 1280 M-groups (2 tiles each)
  const int mT0 = mG * 2;
  const int nT0 = nG * 4;

  const v16bf* aP0 = (const v16bf*)Apack + (size_t)(mT0 + 0) * KT1 * 32 + lane;
  const v16bf* aP1 = (const v16bf*)Apack + (size_t)(mT0 + 1) * KT1 * 32 + lane;
  const v16bf* bP  = (const v16bf*)Bpack + (size_t)nT0 * 32 + lane;

  v8f c[2][4] = {};
#pragma unroll
  for (int kT = 0; kT < KT1; ++kT) {
    const v16bf a0 = aP0[kT * 32];
    const v16bf a1 = aP1[kT * 32];
    v16bf b[4];
#pragma unroll
    for (int j = 0; j < 4; ++j) b[j] = bP[(size_t)(kT * NT1 + j) * 32];
#pragma unroll
    for (int j = 0; j < 4; ++j) {
      c[0][j] = __builtin_amdgcn_wmma_f32_16x16x32_bf16(false, a0, false, b[j],
                                                        (short)0, c[0][j], false, false);
      c[1][j] = __builtin_amdgcn_wmma_f32_16x16x32_bf16(false, a1, false, b[j],
                                                        (short)0, c[1][j], false, false);
    }
  }
  const int nin = lane & 15;
  const int mrb = (lane >> 4) << 3;       // 0 or 8
#pragma unroll
  for (int i = 0; i < 2; ++i) {
#pragma unroll
    for (int j = 0; j < 4; ++j) {
      const int n  = (nT0 + j) * 16 + nin;
      const int mb = (mT0 + i) * 16 + mrb;
      const float bv = bias[n];
#pragma unroll
      for (int r = 0; r < 8; ++r) {
        float v = c[i][j][r] + bv;
        v = v > 0.f ? v : 0.f;
        H[(size_t)(mb + r) * HID + n] = (__bf16)v;
      }
    }
  }
}

// ---------------- aggregation block 1: bf16 gather -> Apack bf16 ------------
__global__ __launch_bounds__(256) void k_agg1(const __bf16* __restrict__ H,
                                              const int* __restrict__ src1,
                                              __bf16* __restrict__ Apack) {
  const int t = threadIdx.x;
  const int rl = t >> 5;                 // 8 rows per block
  const int g  = t & 31;                 // 32 groups of 8 columns
  const int d  = blockIdx.x * 8 + rl;
  const int* s = src1 + d * FANOUT;

  float acc[8];
#pragma unroll
  for (int i = 0; i < 8; ++i) acc[i] = 0.f;
#pragma unroll
  for (int j = 0; j < FANOUT; ++j) {
    const v8bf hv = *(const v8bf*)(H + (size_t)s[j] * HID + g * 8);
#pragma unroll
    for (int i = 0; i < 8; ++i) acc[i] += (float)hv[i];
  }
  v8bf o;
#pragma unroll
  for (int i = 0; i < 8; ++i) o[i] = (__bf16)(acc[i] * 0.1f);

  const int mT  = d >> 4, row = d & 15;
  const int kT  = g >> 2;
  const int kin = (g & 3) * 8;
  const int lane = row + ((kin & 8) ? 16 : 0);
  const int e    = (kin & 16) ? 8 : 0;
  *(v8bf*)(Apack + ((size_t)(mT * KT2 + kT) * 512 + lane * 16 + e)) = o;
}

// --------- GEMM2: Apack x W2 + b2 -> out f32 (1x4 tile blocking) ------------
__global__ __launch_bounds__(256) void k_gemm2(const __bf16* __restrict__ Apack,
                                               const __bf16* __restrict__ Bpack,
                                               const float* __restrict__ bias,
                                               float* __restrict__ out) {
  const int lane = threadIdx.x & 31;
  const int wid  = threadIdx.x >> 5;
  const int mT   = blockIdx.x * 8 + wid;  // 256 M-tiles, wave covers all 4 N-tiles

  const v16bf* aP = (const v16bf*)Apack + (size_t)mT * KT2 * 32 + lane;
  const v16bf* bP = (const v16bf*)Bpack + lane;

  v8f c[4] = {};
#pragma unroll
  for (int kT = 0; kT < KT2; ++kT) {
    const v16bf a = aP[kT * 32];
#pragma unroll
    for (int j = 0; j < 4; ++j) {
      const v16bf b = bP[(size_t)(kT * NT2 + j) * 32];
      c[j] = __builtin_amdgcn_wmma_f32_16x16x32_bf16(false, a, false, b,
                                                     (short)0, c[j], false, false);
    }
  }
  const int nin = lane & 15;
  const int mb  = mT * 16 + ((lane >> 4) << 3);
#pragma unroll
  for (int j = 0; j < 4; ++j) {
    const int n = j * 16 + nin;
    const float bv = bias[n];
#pragma unroll
    for (int r = 0; r < 8; ++r) {
      out[(size_t)(mb + r) * NCLS + n] = c[j][r] + bv;
    }
  }
}

// ---------------------------------------------------------------------------
extern "C" void kernel_launch(void* const* d_in, const int* in_sizes, int n_in,
                              void* d_out, int out_size, void* d_ws, size_t ws_size,
                              hipStream_t stream) {
  (void)in_sizes; (void)n_in; (void)out_size; (void)ws_size;
  const float* feat = (const float*)d_in[0];
  const float* W1   = (const float*)d_in[1];
  const float* b1   = (const float*)d_in[2];
  const float* W2   = (const float*)d_in[3];
  const float* b2   = (const float*)d_in[4];
  const int*   src0 = (const int*)d_in[5];
  // d_in[6] = dst0, d_in[8] = dst1: implicit (edge/FANOUT), unused
  const int*   src1 = (const int*)d_in[7];
  float* out = (float*)d_out;

  // workspace carve-out (256B aligned)
  size_t off = 0;
  auto carve = [&](size_t bytes) {
    void* p = (char*)d_ws + off;
    off += (bytes + 255) & ~(size_t)255;
    return p;
  };
  __bf16* Apack0 = (__bf16*)carve((size_t)GN1 * IN_F * 2);   // ~42 MB
  __bf16* H1     = (__bf16*)carve((size_t)GN1 * HID * 2);    // ~21 MB
  __bf16* Apack1 = (__bf16*)carve((size_t)GN2 * HID * 2);    // ~2 MB
  __bf16* W1p    = (__bf16*)carve((size_t)IN_F * HID * 2);
  __bf16* W2p    = (__bf16*)carve((size_t)HID * NCLS * 2);

  // weight packing (tiny, once per call)
  k_packW<<<(IN_F * HID + 255) / 256, 256, 0, stream>>>(W1, W1p, IN_F, HID);
  k_packW<<<(HID * NCLS + 255) / 256, 256, 0, stream>>>(W2, W2p, HID, NCLS);

  // block 0
  k_agg0<<<GN1 / 4, 256, 0, stream>>>(feat, src0, Apack0);
  k_gemm1<<<(GN1 / 32) * (NT1 / 4) / 8, 256, 0, stream>>>(Apack0, W1p, b1, H1);

  // block 1
  k_agg1<<<GN2 / 8, 256, 0, stream>>>(H1, src1, Apack1);
  k_gemm2<<<(GN2 / 16) / 8, 256, 0, stream>>>(Apack1, W2p, b2, out);
}
